// RNNModel_7499012899021
// MI455X (gfx1250) — matrix operand order repaired and checked by reading, not compiled, not run on hardware
//
#include <hip/hip_runtime.h>
#include <hip/hip_bf16.h>
#include <math.h>

// Problem dims (fixed by reference)
#define Bsz 64
#define Tt  512
#define Dd  768
#define Hh  256
#define Cc  7
#define HS  260   // padded LDS row stride (floats): 4-bank skew -> conflict-free b64 reads

typedef __attribute__((ext_vector_type(16))) __bf16 v16bf;
typedef __attribute__((ext_vector_type(8)))  __bf16 v8bf;
typedef __attribute__((ext_vector_type(4)))  __bf16 v4bf;
typedef __attribute__((ext_vector_type(8)))  float  v8f;
typedef __attribute__((ext_vector_type(2)))  float  v2f;
typedef __attribute__((ext_vector_type(4)))  unsigned int v4u;
typedef __attribute__((ext_vector_type(8)))  int    v8i;
typedef __attribute__((ext_vector_type(4)))  int    v4i;

// gfx1250 feature probes (device pass only; host pass takes fallbacks)
#if defined(__gfx1250__) && __has_builtin(__builtin_amdgcn_tensor_load_to_lds) && \
    __has_builtin(__builtin_amdgcn_s_wait_tensorcnt)
#define USE_TDM 1
#else
#define USE_TDM 0
#endif
#if defined(__gfx1250__) && __has_builtin(__builtin_amdgcn_global_load_async_to_lds_b128) && \
    __has_builtin(__builtin_amdgcn_s_wait_asynccnt)
#define USE_ASYNC 1
#else
#define USE_ASYNC 0
#endif

// LDS layout (floats) for rnn_scan
#define WHH_F   0
#define H_F     (256 * HS)                  // 66560
#define WCLS_F  (H_F + 16 * HS)             // 70720
#define BCLS_F  (WCLS_F + Cc * Hh)          // 72512
#define XIB0_F  (BCLS_F + 8)                // 72520
#define XIB1_F  (XIB0_F + 16 * HS)          // 76680
#define SMEM_F  (XIB1_F + 16 * HS)          // 80840 floats = 323,360 B <= 320 KB

// ---------------------------------------------------------------------------
// Kernel 0: fp32 -> bf16 conversion (x and W_ih). Output stays L2-resident.
// ---------------------------------------------------------------------------
__global__ void cvt_f32_to_bf16(const float* __restrict__ in,
                                __bf16* __restrict__ out, int n) {
    int i = (blockIdx.x * blockDim.x + threadIdx.x) * 4;
    if (i + 3 < n) {
        float4 v = *(const float4*)(in + i);
        v4bf o;
        o[0] = (__bf16)v.x; o[1] = (__bf16)v.y;
        o[2] = (__bf16)v.z; o[3] = (__bf16)v.w;
        *(v4bf*)(out + i) = o;
    } else {
        for (; i < n; ++i) out[i] = (__bf16)in[i];
    }
}

// ---------------------------------------------------------------------------
// Kernel 1: input projection  xi[t,b,h] = sum_d x[b,t,d]*W_ih[h,d] + bih[h]+bhh[h]
// One wave per (t, n-tile); 4 M-subtiles reuse the B operand. bf16 WMMA, f32 acc.
// ---------------------------------------------------------------------------
__global__ void input_gemm(const __bf16* __restrict__ xbf,   // [B,T,D] bf16
                           const __bf16* __restrict__ wbf,   // [H,D]  bf16
                           const float*  __restrict__ b_ih,
                           const float*  __restrict__ b_hh,
                           float* __restrict__ xi) {         // [T,B,H] fp32
    const int gw   = (blockIdx.x * blockDim.x + threadIdx.x) >> 5;  // 0..8191
    const int lane = threadIdx.x & 31;
    const int t    = gw >> 4;
    const int nt   = gw & 15;
    const int half = lane >> 4;
    const int lrow = lane & 15;
    const int col  = nt * 16 + lrow;

    const __bf16* brow = wbf + (size_t)col * Dd;          // W_ih row n (= B column)
    const __bf16* arow[4];
#pragma unroll
    for (int s = 0; s < 4; ++s)
        arow[s] = xbf + ((size_t)(16 * s + lrow) * Tt + t) * Dd + 8 * half;

    v8f acc[4] = {v8f{}, v8f{}, v8f{}, v8f{}};

    for (int k0 = 0; k0 < Dd; k0 += 32) {
        v8bf blo = *(const v8bf*)(brow + k0 + 8 * half);
        v8bf bhi = *(const v8bf*)(brow + k0 + 8 * half + 16);
        v16bf bm = __builtin_shufflevector(blo, bhi,
                     0,1,2,3,4,5,6,7,8,9,10,11,12,13,14,15);
#pragma unroll
        for (int s = 0; s < 4; ++s) {
            v8bf alo = *(const v8bf*)(arow[s] + k0);        // K run 1
            v8bf ahi = *(const v8bf*)(arow[s] + k0 + 16);   // K run 2 (+16)
            v16bf am = __builtin_shufflevector(alo, ahi,
                         0,1,2,3,4,5,6,7,8,9,10,11,12,13,14,15);
            acc[s] = __builtin_amdgcn_wmma_f32_16x16x32_bf16(
                         false, am, false, bm, (short)0, acc[s], false, false);
        }
    }

    const float bias = b_ih[col] + b_hh[col];
#pragma unroll
    for (int s = 0; s < 4; ++s) {
#pragma unroll
        for (int r = 0; r < 8; ++r) {
            int row = t * Bsz + 16 * s + 8 * half + r;     // [T,B] row
            xi[(size_t)row * Hh + col] = acc[s][r] + bias;
        }
    }
}

#if USE_ASYNC
typedef __attribute__((address_space(1))) v4i gv4i;   // global int4
typedef __attribute__((address_space(3))) v4i lv4i;   // LDS int4

// Issue the 16 KB xi tile for step `tstep` into the LDS buffer at bufFloatOff.
// 1024 16-byte chunks / 512 threads = 2 async b128 per lane (ASYNCcnt +2 / wave).
__device__ __forceinline__ void async_copy_xi(const float* __restrict__ xi,
                                              float* smem, int bufFloatOff,
                                              int tstep, int b0, int tid) {
#pragma unroll
    for (int j = 0; j < 2; ++j) {
        int q   = tid * 2 + j;
        int row = q >> 6;                 // 0..15
        int cq  = q & 63;                 // 16-byte chunk within row
        const float* src = xi + ((size_t)tstep * Bsz + b0 + row) * Hh + cq * 4;
        float*       dst = smem + bufFloatOff + row * HS + cq * 4;
        __builtin_amdgcn_global_load_async_to_lds_b128(
            (gv4i*)(void*)src, (lv4i*)(void*)dst, 0, 0);
    }
}
#endif

// ---------------------------------------------------------------------------
// Kernel 2: sequential scan + fused classifier.
// 4 blocks x 512 threads (16 waves). W_hh staged by TDM with pad_interval/amount
// giving the 260-float bank-skewed rows directly; xi double-buffered in LDS by
// async b128 copies overlapped with the fp32 WMMA K-loop.
// ---------------------------------------------------------------------------
__global__ void rnn_scan(const float* __restrict__ xi,     // [T,B,H]
                         const float* __restrict__ W_hh,   // [H,H]
                         const float* __restrict__ W_cls,  // [C,H]
                         const float* __restrict__ b_cls,  // [C]
                         float* __restrict__ out) {        // [B,T,C]
    extern __shared__ float smem[];
    float* whh_s  = smem + WHH_F;
    float* h_s    = smem + H_F;
    float* wcls_s = smem + WCLS_F;
    float* bcls_s = smem + BCLS_F;

    const int tid = threadIdx.x;
    const int b0  = blockIdx.x * 16;

#if USE_TDM
    if (tid < 32) {   // wave 0 issues one TDM descriptor for all of W_hh
        unsigned long long ga = (unsigned long long)(const void*)W_hh;
        v4u g0;
        g0.x = 1u;                                          // count=1, user mode
        g0.y = (unsigned)(WHH_F * 4);                       // lds_addr (bytes)
        g0.z = (unsigned)(ga & 0xffffffffu);                // global_addr[31:0]
        g0.w = (unsigned)((ga >> 32) & 0x01ffffffu) | (2u << 30);  // [56:32], type=2
        v8i g1;
        g1[0] = (int)((2u << 16)      // data_size = 4B
                    | (1u << 20)      // pad_enable
                    | (7u << 22)      // pad_interval: 256 DWORDs
                    | (3u << 25));    // pad_amount:   4 DWORDs  -> 260-float rows
        g1[1] = (int)(256u << 16);    // tensor_dim0 = 256 (low 16 bits at [31:16])
        g1[2] = (int)(256u << 16);    // tensor_dim1 = 256
        g1[3] = (int)(256u << 16);    // tile_dim0   = 256
        g1[4] = 256;                  // tile_dim1   = 256
        g1[5] = 256;                  // tensor_dim0_stride = 256
        g1[6] = 0;
        g1[7] = 0;
        v4i gz4 = {0, 0, 0, 0};
        v8i gz8 = {0, 0, 0, 0, 0, 0, 0, 0};
        // 6-arg form (clang-23 / therock HIP headers): groups 2/3 + extra group
        // are zero descriptors for a plain 2D tensor.
        __builtin_amdgcn_tensor_load_to_lds(g0, g1, gz4, gz4, gz8, 0);
        __builtin_amdgcn_s_wait_tensorcnt(0);
    }
#else
    for (int i = tid; i < Hh * Hh; i += 512)
        whh_s[(i >> 8) * HS + (i & 255)] = W_hh[i];
#endif
    for (int i = tid; i < Cc * Hh; i += 512) wcls_s[i] = W_cls[i];
    if (tid < Cc) bcls_s[tid] = b_cls[tid];
    for (int i = tid; i < 16 * HS; i += 512) h_s[i] = 0.f;   // h0 = zeros
#if USE_ASYNC
    async_copy_xi(xi, smem, XIB0_F, 0, b0, tid);             // preload xi[0]
#endif
    __syncthreads();

    const int w    = tid >> 5;
    const int lane = tid & 31;
    const int half = lane >> 4;
    const int lrow = lane & 15;
    const float* ha = h_s + lrow * HS + 2 * half;              // A: h_prev row
    const float* ba = whh_s + (w * 16 + lrow) * HS + 2 * half; // B: W_hh row n
    const int colD = w * 16 + lrow;                            // D column
    const int mD   = 8 * half;                                 // D row base

    for (int t = 0; t < Tt; ++t) {
#if USE_ASYNC
        if (t + 1 < Tt)   // overlap next step's copy with this step's WMMA
            async_copy_xi(xi, smem, ((t + 1) & 1) ? XIB1_F : XIB0_F, t + 1, b0, tid);
#endif
        // ---- h_prev @ W_hh^T for this wave's 16x16 tile (fp32 WMMA) ----
        v8f acc = v8f{};
#pragma unroll 8
        for (int k0 = 0; k0 < Hh; k0 += 4) {
            v2f a = *(const v2f*)(ha + k0);   // ds_load_b64, conflict-free (HS=260)
            v2f b = *(const v2f*)(ba + k0);
            acc = __builtin_amdgcn_wmma_f32_16x16x4_f32(
                      false, a, false, b, (short)0, acc, false, false);
        }

        float hv[8];
#if USE_ASYNC
        if (t + 1 < Tt) __builtin_amdgcn_s_wait_asynccnt(2); // retire step-t copies only
        else            __builtin_amdgcn_s_wait_asynccnt(0);
        __syncthreads();            // async data visible to all waves; h_s reads done
        {
            const float* xis = smem + ((t & 1) ? XIB1_F : XIB0_F);
#pragma unroll
            for (int r = 0; r < 8; ++r)
                hv[r] = tanhf(acc[r] + xis[(mD + r) * HS + colD]);
        }
#else
        const float* xit = xi + ((size_t)t * Bsz + b0) * Hh;
        if (t + 1 < Tt)
            __builtin_prefetch(xi + ((size_t)(t + 1) * Bsz + b0) * Hh + tid * 8, 0, 1);
#pragma unroll
        for (int r = 0; r < 8; ++r)
            hv[r] = tanhf(acc[r] + xit[(mD + r) * Hh + colD]);
        __syncthreads();            // all K-loop reads of h_s complete
#endif
#pragma unroll
        for (int r = 0; r < 8; ++r)
            h_s[(mD + r) * HS + colD] = hv[r]; // consecutive cols -> no conflicts
        __syncthreads();                       // h_new visible

        // ---- fused classifier: out[b,t,c] = relu(h) . W_cls[c] + b_cls[c] ----
        if (tid < 448) {                       // 14 full waves; groups of 4 lanes
            int oid = tid >> 2, part = tid & 3;
            int bb = oid / Cc, c = oid - bb * Cc;
            const float* hr = h_s + bb * HS + part * 64;
            const float* wr = wcls_s + c * Hh + part * 64;
            float s = 0.f;
#pragma unroll 8
            for (int k = 0; k < 64; ++k)
                s = fmaf(fmaxf(hr[k], 0.f), wr[k], s);
            s += __shfl_xor(s, 1, 32);
            s += __shfl_xor(s, 2, 32);
            if (part == 0)
                out[(((size_t)(b0 + bb)) * Tt + t) * Cc + c] = s + bcls_s[c];
        }
    }
}

// ---------------------------------------------------------------------------
// Launch
// ---------------------------------------------------------------------------
extern "C" void kernel_launch(void* const* d_in, const int* in_sizes, int n_in,
                              void* d_out, int out_size, void* d_ws, size_t ws_size,
                              hipStream_t stream) {
    (void)in_sizes; (void)n_in; (void)out_size; (void)ws_size;
    const float* x     = (const float*)d_in[0];
    const float* W_ih  = (const float*)d_in[1];
    const float* b_ih  = (const float*)d_in[2];
    const float* W_hh  = (const float*)d_in[3];
    const float* b_hh  = (const float*)d_in[4];
    const float* W_cls = (const float*)d_in[5];
    const float* b_cls = (const float*)d_in[6];
    float* out = (float*)d_out;

    // Workspace layout (~80.4 MB): xi fp32 | x bf16 | W_ih bf16
    char*   ws  = (char*)d_ws;
    float*  xi  = (float*)ws;                                   // T*B*H*4 = 32 MB
    __bf16* xbf = (__bf16*)(ws + (size_t)Tt * Bsz * Hh * 4);    // B*T*D*2 = 48 MB
    __bf16* wbf = (__bf16*)(ws + (size_t)Tt * Bsz * Hh * 4
                               + (size_t)Bsz * Tt * Dd * 2);    // H*D*2

    const int nx = Bsz * Tt * Dd;   // 25,165,824 (mult of 4)
    const int nw = Hh * Dd;         // 196,608    (mult of 4)
    cvt_f32_to_bf16<<<(nx / 4 + 255) / 256, 256, 0, stream>>>(x, xbf, nx);
    cvt_f32_to_bf16<<<(nw / 4 + 255) / 256, 256, 0, stream>>>(W_ih, wbf, nw);

    // 8192 wave-tasks (t x 16 n-tiles), 8 waves/block of 256 threads
    input_gemm<<<1024, 256, 0, stream>>>(xbf, wbf, b_ih, b_hh, xi);

    const size_t smem_bytes = (size_t)SMEM_F * 4;   // 323,360 B <= 320 KB WGP LDS
    rnn_scan<<<4, 512, smem_bytes, stream>>>(xi, W_hh, W_cls, b_cls, out);
}
